// CausalSelfAttention_6777458393858
// MI455X (gfx1250) — matrix operand order repaired and checked by reading, not compiled, not run on hardware
//
#include <hip/hip_runtime.h>

// ---------------------------------------------------------------------------
// CDNA5 (gfx1250) causal self-attention forward, all matmuls via
// v_wmma_f32_16x16x32_bf16 (wave32). Flash-attention K tiles staged through
// LDS with global_load_async_to_lds_b128 (double buffered) when available.
// ---------------------------------------------------------------------------

typedef __attribute__((ext_vector_type(16))) __bf16        v16bf;
typedef __attribute__((ext_vector_type(8)))  float         v8f;
typedef __attribute__((ext_vector_type(8)))  unsigned short us8;
typedef __attribute__((ext_vector_type(4)))  int           v4i;
typedef __attribute__((address_space(1))) v4i gv4i;   // global int4
typedef __attribute__((address_space(3))) v4i lv4i;   // LDS int4

#if __has_builtin(__builtin_amdgcn_global_load_async_to_lds_b128) && \
    __has_builtin(__builtin_amdgcn_s_wait_asynccnt)
#define USE_ASYNC_LDS 1
#else
#define USE_ASYNC_LDS 0
#endif

__device__ __forceinline__ unsigned short f2bf(float f) {
  unsigned int u = __float_as_uint(f);
  u += 0x7FFFu + ((u >> 16) & 1u);          // round-to-nearest-even
  return (unsigned short)(u >> 16);
}

// A-matrix 16x32 bf16 fragment: lane (m = lane&15); half 0 holds K in
// {k0..k0+7, k0+16..k0+23}, half 1 holds {k0+8.., k0+24..} (ISA 7.12.2).
__device__ __forceinline__ v16bf load_fragA(const unsigned short* row, int k0, int hv) {
  union { us8 u[2]; v16bf v; } t;
  const unsigned short* p = row + k0 + hv * 8;
  t.u[0] = *(const us8*)(p);
  t.u[1] = *(const us8*)(p + 16);
  return t.v;
}

// B-matrix 32x16 bf16 fragment: lane holds column n = lane&15; half 0 holds
// K in {k0..k0+7, k0+16..k0+23}, half 1 holds {k0+8.., k0+24..} (same as A).
__device__ __forceinline__ v16bf load_fragB(const unsigned short* row, int k0, int hv) {
  union { us8 u[2]; v16bf v; } t;
  const unsigned short* p = row + k0 + hv * 8;
  t.u[0] = *(const us8*)(p);
  t.u[1] = *(const us8*)(p + 16);
  return t.v;
}

__device__ __forceinline__ v8f wmma_bf16(v16bf a, v16bf b, v8f c) {
  return __builtin_amdgcn_wmma_f32_16x16x32_bf16(false, a, false, b, (short)0, c,
                                                 false, false);
}

__device__ __forceinline__ float redmax16(float v) {
  v = fmaxf(v, __shfl_xor(v, 1));
  v = fmaxf(v, __shfl_xor(v, 2));
  v = fmaxf(v, __shfl_xor(v, 4));
  v = fmaxf(v, __shfl_xor(v, 8));
  return v;
}
__device__ __forceinline__ float redsum16(float v) {
  v += __shfl_xor(v, 1);
  v += __shfl_xor(v, 2);
  v += __shfl_xor(v, 4);
  v += __shfl_xor(v, 8);
  return v;
}

// ---------------------------------------------------------------------------
// fp32 -> bf16 convert
// ---------------------------------------------------------------------------
__global__ void cvt_bf16_kernel(const float* __restrict__ src,
                                unsigned short* __restrict__ dst, int n) {
  for (int i = blockIdx.x * blockDim.x + threadIdx.x; i < n;
       i += gridDim.x * blockDim.x)
    dst[i] = f2bf(src[i]);
}

// ---------------------------------------------------------------------------
// y = A(bf16, MxK row-major) @ Bw(bf16, NxK row-major)^T
// Block: 256 thr = 8 waves, each wave = 32(M) x 64(N) tile; K-step 32;
// B fragments reused across both M sub-tiles (8 WMMAs / step / wave).
// mode 0: plain fp32 store to out_f (MxN)
// mode 1: fused RoPE (rotate-half) + bf16 store to (B,H,T,64)
// mode 2: bf16 store transposed to (B,H,64,T)    (for V^T)
// ---------------------------------------------------------------------------
__global__ __launch_bounds__(256) void gemm_bf16_kernel(
    const unsigned short* __restrict__ A, const unsigned short* __restrict__ Bw,
    int M, int N, int K, int mode, const float* __restrict__ cosp,
    const float* __restrict__ sinp, unsigned short* __restrict__ out_bf,
    float* __restrict__ out_f, int T, int H) {
  const int lane = threadIdx.x & 31;
  const int wid  = threadIdx.x >> 5;
  const int hv   = lane >> 4;
  const int ln   = lane & 15;
  const int m_base = blockIdx.x * 256 + wid * 32;
  const int n_base = blockIdx.y * 64;

  const unsigned short* Arow0 = A  + (size_t)(m_base + ln) * K;
  const unsigned short* Arow1 = A  + (size_t)(m_base + 16 + ln) * K;
  const unsigned short* Brow  = Bw + (size_t)(n_base + ln) * K;

  const v8f z = {0.f, 0.f, 0.f, 0.f, 0.f, 0.f, 0.f, 0.f};
  v8f acc[2][4] = {{z, z, z, z}, {z, z, z, z}};

  for (int k0 = 0; k0 < K; k0 += 32) {
    // prefetch the streaming A operand a few tiles ahead
    __builtin_prefetch(Arow0 + k0 + 512, 0, 0);
    __builtin_prefetch(Arow1 + k0 + 512, 0, 0);
    v16bf a0 = load_fragA(Arow0, k0, hv);
    v16bf a1 = load_fragA(Arow1, k0, hv);
#pragma unroll
    for (int c = 0; c < 4; ++c) {
      v16bf bb = load_fragB(Brow + (size_t)(16 * c) * K, k0, hv);
      acc[0][c] = wmma_bf16(a0, bb, acc[0][c]);
      acc[1][c] = wmma_bf16(a1, bb, acc[1][c]);
    }
  }

  if (mode == 0) {
#pragma unroll
    for (int ms = 0; ms < 2; ++ms)
#pragma unroll
      for (int c = 0; c < 4; ++c)
#pragma unroll
        for (int r = 0; r < 8; ++r)
          out_f[(size_t)(m_base + 16 * ms + r + hv * 8) * N +
                (n_base + 16 * c + ln)] = acc[ms][c][r];
  } else {
    const int h = n_base >> 6;  // N-tile (64) is head-aligned
#pragma unroll
    for (int ms = 0; ms < 2; ++ms) {
#pragma unroll
      for (int r = 0; r < 8; ++r) {
        const int mg = m_base + 16 * ms + r + hv * 8;
        const int b  = mg / T;
        const int t  = mg % T;
        const size_t qbase = ((size_t)(b * H + h) * T + t) * 64;
#pragma unroll
        for (int c = 0; c < 4; ++c) {
          const int dd = 16 * c + ln;
          float val = acc[ms][c][r];
          if (mode == 1) {
            // rotate-half pair lives on the same lane, chunk c +/- 2
            const float pr = (c < 2) ? -acc[ms][c + 2][r] : acc[ms][c - 2][r];
            val = val * cosp[t * 64 + dd] + pr * sinp[t * 64 + dd];
            out_bf[qbase + dd] = f2bf(val);
          } else {  // mode 2: V^T  (B,H,64,T)
            out_bf[((size_t)(b * H + h) * 64 + dd) * (size_t)T + t] = f2bf(val);
          }
        }
      }
    }
  }
}

// ---------------------------------------------------------------------------
// Flash attention: one wave per 16-row q tile per (b,h). Online softmax,
// causal + key-padding mask, 32 keys per step (8 WMMAs / step). K tiles are
// staged into wave-private LDS with async loads (double buffered).
// ---------------------------------------------------------------------------
#define FA_WAVES 8
__global__ __launch_bounds__(256) void flash_attn_kernel(
    const unsigned short* __restrict__ Q, const unsigned short* __restrict__ Kr,
    const unsigned short* __restrict__ VT, const int* __restrict__ amask,
    unsigned short* __restrict__ att, int Bn, int Hh, int T) {
  __shared__ __align__(16) unsigned short plds[FA_WAVES][16 * 48];
#if USE_ASYNC_LDS
  __shared__ __align__(16) unsigned short kstage[FA_WAVES][2][32 * 64];
#endif

  const int lane = threadIdx.x & 31;
  const int wid  = threadIdx.x >> 5;
  const int hv   = lane >> 4;
  const int ln   = lane & 15;

  const int ntq  = T >> 4;
  const int tile = blockIdx.x * FA_WAVES + wid;
  const int qt   = tile % ntq;
  const int bh   = tile / ntq;
  const int b    = bh / Hh;
  const int h    = bh % Hh;
  const int t0   = qt << 4;

  const size_t bhT = (size_t)bh * T;
  const unsigned short* qrow = Q + (bhT + (size_t)(t0 + ln)) * 64;
  const v16bf qa0 = load_fragA(qrow, 0, hv);
  const v16bf qa1 = load_fragA(qrow, 32, hv);
  const int* maskb = amask + (size_t)b * T;

  const v8f z = {0.f, 0.f, 0.f, 0.f, 0.f, 0.f, 0.f, 0.f};
  v8f O0 = z, O1 = z, O2 = z, O3 = z;
  float m[8], l[8];
#pragma unroll
  for (int r = 0; r < 8; ++r) { m[r] = -3.0e38f; l[r] = 0.f; }

  unsigned short* pl = &plds[wid][0];

#if USE_ASYNC_LDS
  // Stage one 32x64 bf16 K tile (contiguous 4KB) into LDS: 8 async b128/wave.
  auto stageK = [&](int j0, int buf) {
    const char* gbase = (const char*)(Kr + (bhT + (size_t)j0) * 64);
    char* lbase = (char*)&kstage[wid][buf][0];
#pragma unroll
    for (int i = 0; i < 8; ++i) {
      const int q16 = (lane + 32 * i) * 16;  // 256 chunks of 16B
      __builtin_amdgcn_global_load_async_to_lds_b128(
          (gv4i*)(gbase + q16), (lv4i*)(lbase + q16), 0, 0);
    }
  };
  stageK(0, 0);
#endif

  const int nsteps = (t0 + 16 + 31) >> 5;
  for (int s = 0; s < nsteps; ++s) {
    const int j0 = s << 5;
    // ---- S = (q @ k^T) for 32 key columns (two 16-wide C tiles) ----
    v8f Sa = z, Sb = z;
    {
#if USE_ASYNC_LDS
      // Prefetch next tile, then wait so the current tile (issued earlier,
      // completes in order) is resident. Overshoot stays inside d_ws.
      stageK(j0 + 32, (s + 1) & 1);
      __builtin_amdgcn_s_wait_asynccnt(8);
      const unsigned short* k0r = &kstage[wid][s & 1][ln * 64];
#else
      const unsigned short* k0r = Kr + (bhT + (size_t)(j0 + ln)) * 64;
#endif
      const unsigned short* k1r = k0r + 16 * 64;
      Sa = wmma_bf16(qa0, load_fragB(k0r, 0,  hv), Sa);
      Sa = wmma_bf16(qa1, load_fragB(k0r, 32, hv), Sa);
      Sb = wmma_bf16(qa0, load_fragB(k1r, 0,  hv), Sb);
      Sb = wmma_bf16(qa1, load_fragB(k1r, 32, hv), Sb);
    }

    const int key0 = j0 + ln;
    const int key1 = key0 + 16;
    const bool km0 = maskb[key0] != 0;
    const bool km1 = maskb[key1] != 0;

    // ---- masked online softmax, write P (bf16) to LDS in A layout ----
#pragma unroll
    for (int r = 0; r < 8; ++r) {
      const int tr = t0 + r + hv * 8;
      float s0 = (km0 && key0 <= tr) ? Sa[r] * 0.125f : -1.0e30f;
      float s1 = (km1 && key1 <= tr) ? Sb[r] * 0.125f : -1.0e30f;
      const float rowmax = redmax16(fmaxf(s0, s1));
      const float mn = fmaxf(m[r], rowmax);
      const float p0 = __expf(s0 - mn);
      const float p1 = __expf(s1 - mn);
      const float al = __expf(m[r] - mn);
      m[r] = mn;
      l[r] = l[r] * al + redsum16(p0 + p1);
      O0[r] *= al; O1[r] *= al; O2[r] *= al; O3[r] *= al;
      const int ro = (r + hv * 8) * 48;
      pl[ro + ln]      = f2bf(p0);
      pl[ro + 16 + ln] = f2bf(p1);
    }
    asm volatile("s_wait_dscnt 0x0" ::: "memory");  // LDS P visible to loads

    // ---- O += P @ V   (V stored transposed: rows = head-dim) ----
    const v16bf pf = load_fragA(pl + ln * 48, 0, hv);
    const unsigned short* vrow =
        VT + ((size_t)bh * 64 + (size_t)ln) * (size_t)T + j0;
    O0 = wmma_bf16(pf, load_fragB(vrow,                   0, hv), O0);
    O1 = wmma_bf16(pf, load_fragB(vrow + (size_t)16 * T,  0, hv), O1);
    O2 = wmma_bf16(pf, load_fragB(vrow + (size_t)32 * T,  0, hv), O2);
    O3 = wmma_bf16(pf, load_fragB(vrow + (size_t)48 * T,  0, hv), O3);
  }

  // ---- normalize and store (B,T,D) bf16 (pre output-projection) ----
  const int Dt = Hh * 64;
#pragma unroll
  for (int r = 0; r < 8; ++r) {
    const float rl = 1.0f / l[r];
    const int t = t0 + r + hv * 8;
    const size_t ob = ((size_t)b * T + t) * (size_t)Dt + (size_t)h * 64 + ln;
    att[ob]      = f2bf(O0[r] * rl);
    att[ob + 16] = f2bf(O1[r] * rl);
    att[ob + 32] = f2bf(O2[r] * rl);
    att[ob + 48] = f2bf(O3[r] * rl);
  }
}

// ---------------------------------------------------------------------------
extern "C" void kernel_launch(void* const* d_in, const int* in_sizes, int n_in,
                              void* d_out, int out_size, void* d_ws,
                              size_t ws_size, hipStream_t stream) {
  (void)in_sizes; (void)n_in; (void)out_size; (void)ws_size;
  const int B = 4, T = 2048, D = 1024, H = 16;
  const int M = B * T;  // 8192

  const float* x     = (const float*)d_in[0];
  const int*   amask = (const int*)d_in[1];
  const float* cosp  = (const float*)d_in[2];
  const float* sinp  = (const float*)d_in[3];
  const float* Wq    = (const float*)d_in[4];
  const float* Wk    = (const float*)d_in[5];
  const float* Wv    = (const float*)d_in[6];
  const float* Wo    = (const float*)d_in[7];

  const size_t MTD = (size_t)M * D;   // 8,388,608
  const size_t DDs = (size_t)D * D;   // 1,048,576
  unsigned short* xb  = (unsigned short*)d_ws;
  unsigned short* wqb = xb  + MTD;
  unsigned short* wkb = wqb + DDs;
  unsigned short* wvb = wkb + DDs;
  unsigned short* wob = wvb + DDs;
  unsigned short* qr  = wob + DDs;    // (B,H,T,64) bf16, rope applied
  unsigned short* kr  = qr  + MTD;    // (B,H,T,64) bf16, rope applied
  unsigned short* vT  = kr  + MTD;    // (B,H,64,T) bf16  (also async overshoot pad)
  unsigned short* att = vT  + MTD;    // (B,T,D)   bf16, pre-Wo

  // 1) fp32 -> bf16
  cvt_bf16_kernel<<<4096, 256, 0, stream>>>(x,  xb,  (int)MTD);
  cvt_bf16_kernel<<<1024, 256, 0, stream>>>(Wq, wqb, (int)DDs);
  cvt_bf16_kernel<<<1024, 256, 0, stream>>>(Wk, wkb, (int)DDs);
  cvt_bf16_kernel<<<1024, 256, 0, stream>>>(Wv, wvb, (int)DDs);
  cvt_bf16_kernel<<<1024, 256, 0, stream>>>(Wo, wob, (int)DDs);

  // 2) Q/K/V projections (RoPE fused into Q,K epilogue; V stored transposed)
  dim3 gg(M / 256, D / 64);
  gemm_bf16_kernel<<<gg, 256, 0, stream>>>(xb, wqb, M, D, D, 1, cosp, sinp, qr,
                                           nullptr, T, H);
  gemm_bf16_kernel<<<gg, 256, 0, stream>>>(xb, wkb, M, D, D, 1, cosp, sinp, kr,
                                           nullptr, T, H);
  gemm_bf16_kernel<<<gg, 256, 0, stream>>>(xb, wvb, M, D, D, 2, nullptr,
                                           nullptr, vT, nullptr, T, H);

  // 3) flash attention (causal + key padding mask)
  const int ntiles = B * H * (T / 16);           // 8192
  flash_attn_kernel<<<ntiles / FA_WAVES, 256, 0, stream>>>(qr, kr, vT, amask,
                                                           att, B, H, T);

  // 4) output projection -> fp32 d_out
  gemm_bf16_kernel<<<gg, 256, 0, stream>>>(att, wob, M, D, D, 0, nullptr,
                                           nullptr, nullptr, (float*)d_out, T,
                                           H);
}
